// CrowdsClassificationCModel_80650895884840
// MI455X (gfx1250) — matrix operand order repaired and checked by reading, not compiled, not run on hardware
//
#include <hip/hip_runtime.h>

// CrowdsClassification fused kernel for MI455X (gfx1250, wave32, WMMA).
// out[b,d,r] = sum_c p[b,c] * softmax_d( x[b,:]@W[:, (c*8+d)*64+r] + bias[c,d,r] )
//
// GEMM (16384 x 4096 x 128) in f16 WMMA with f32 accumulation
// (v_wmma_f32_16x16x32_f16). Two 16-row M-tiles per workgroup so every
// B fragment load feeds two WMMAs (1:1 b128-load : wmma ratio). B fragments
// are prefetched into a register array so loads clause ahead of the WMMAs.
// Softmax + p-contraction fused through a 128 KB LDS logits tile.

typedef __attribute__((ext_vector_type(16))) _Float16 v16h;
typedef __attribute__((ext_vector_type(8)))  float    v8f;

#define BDIM   16384
#define FDIM   128
#define CCLS   8
#define RDIM   64
#define NKDIM  64            // C*C
#define NCOLS  4096          // NKDIM * RDIM
#define BT     32            // b rows per workgroup (2 WMMA M-tiles)
#define RT     16            // r-tile (WMMA N)
#define NWAVES 8             // 256 threads
#define KPW    (NKDIM / NWAVES)   // 8 k-values per wave

// LDS logits tile: [k(64)][r(16)][b(32)] f32 = 128 KB.
// Swizzle b' = (b + 2r) & 31 keeps both the WMMA C-layout stores
// (lanes sweep r at fixed b) and the phase-2 reads conflict-free.
#define LSWZ(r, b) (((r) * BT) + (((b) + 2 * (r)) & 31))

// ---------------------------------------------------------------------------
// Prep: W (f32, row-major F x NCOLS) -> Wt (f16, column-major: Wt[n][k]).
// Makes the per-wave WMMA B-fragment two contiguous 16B loads per K-chunk.
// ---------------------------------------------------------------------------
__global__ __launch_bounds__(256) void conv_w_kernel(const float* __restrict__ W,
                                                     _Float16* __restrict__ Wt) {
  int idx = blockIdx.x * 256 + threadIdx.x;   // 0 .. F*NCOLS-1 (coalesced read)
  int k = idx >> 12;                          // / NCOLS
  int n = idx & (NCOLS - 1);
  Wt[(size_t)n * FDIM + k] = (_Float16)W[idx];
}

// ---------------------------------------------------------------------------
// Fused GEMM + bias + row-softmax + p-contraction.
// Grid: (RDIM/RT, BDIM/BT) = (4, 512). Block: 256 threads = 8 waves.
// Wave w computes logits for k in [8w, 8w+8) over a 32b x 16r tile.
// ---------------------------------------------------------------------------
__global__ __launch_bounds__(256) void fused_kernel(
    const float* __restrict__ x, const float* __restrict__ p,
    const _Float16* __restrict__ Wt, const float* __restrict__ bias,
    float* __restrict__ out) {
  __shared__ float lds[NKDIM * RT * BT];   // 131072 bytes

  const int tid  = threadIdx.x;
  const int wave = tid >> 5;
  const int lane = tid & 31;
  const int h    = lane >> 4;   // lane half
  const int ln   = lane & 15;

  const int r0 = blockIdx.x * RT;
  const int b0 = blockIdx.y * BT;
  const int k0 = wave * KPW;

  // ---- A fragments: x[b0 + t*16 + m, 0..127] f32 -> f16, WMMA A 16x32 layout.
  // Lane (half h, m=ln) holds K = kc + 8h + e (e<8) and kc + 16 + 8h + (e-8).
  v16h A[2][4];
  #pragma unroll
  for (int t = 0; t < 2; ++t) {
    const float* xrow = x + (size_t)(b0 + t * 16 + ln) * FDIM + 8 * h;
    #pragma unroll
    for (int cch = 0; cch < 4; ++cch) {
      float tmp[16] __attribute__((aligned(16)));
      *(float4*)(tmp + 0)  = *(const float4*)(xrow + cch * 32 + 0);
      *(float4*)(tmp + 4)  = *(const float4*)(xrow + cch * 32 + 4);
      *(float4*)(tmp + 8)  = *(const float4*)(xrow + cch * 32 + 16);
      *(float4*)(tmp + 12) = *(const float4*)(xrow + cch * 32 + 20);
      #pragma unroll
      for (int e = 0; e < 16; ++e) A[t][cch][e] = (_Float16)tmp[e];
    }
  }

  // ---- Precompute swizzled LDS store indices (k-invariant part).
  // Stores per kt only add kt*RT*BT floats = kt*2048 bytes (fits ds imm).
  int sidx[16];
  #pragma unroll
  for (int i = 0; i < 8; ++i) {
    sidx[i]     = k0 * (RT * BT) + LSWZ(ln, i + 8 * h);        // tile 0
    sidx[i + 8] = k0 * (RT * BT) + LSWZ(ln, 16 + i + 8 * h);   // tile 1
  }

  // ---- GEMM: 8 k-tiles per wave; each B fragment feeds 2 WMMAs (M=32).
  #pragma unroll
  for (int kt = 0; kt < KPW; ++kt) {
    const int k   = k0 + kt;
    const int col = k * RDIM + r0 + ln;                       // B column n
    const _Float16* wp = Wt + (size_t)col * FDIM + 8 * h;     // K = 8h + e, 16 + 8h + (e-8)

    // Prefetch all 4 B chunks into distinct registers -> one load clause,
    // WMMAs overlap the tail of the loads.
    v16h Bm[4];
    #pragma unroll
    for (int cch = 0; cch < 4; ++cch) {
      *((float4*)&Bm[cch])     = *(const float4*)(wp + cch * 32);
      *((float4*)&Bm[cch] + 1) = *(const float4*)(wp + cch * 32 + 16);
    }
    const float bv = bias[k * RDIM + r0 + ln];   // column bias

    v8f acc0 = {};
    v8f acc1 = {};
    #pragma unroll
    for (int cch = 0; cch < 4; ++cch) {
      acc0 = __builtin_amdgcn_wmma_f32_16x16x32_f16(
          false, A[0][cch], false, Bm[cch], (short)0, acc0, false, false);
      acc1 = __builtin_amdgcn_wmma_f32_16x16x32_f16(
          false, A[1][cch], false, Bm[cch], (short)0, acc1, false, false);
    }

    #pragma unroll
    for (int i = 0; i < 8; ++i) {
      // C/D layout: VGPR i -> row M = i + 8h, column N = ln.
      lds[sidx[i]     + kt * (RT * BT)] = acc0[i] + bv;
      lds[sidx[i + 8] + kt * (RT * BT)] = acc1[i] + bv;
    }
  }
  __syncthreads();

  // ---- Row softmax over d (8 values) + contraction with p over c.
  for (int q = tid; q < BT * RT; q += 256) {
    const int r  = q & 15;        // lanes sweep r -> coalesced output stores
    const int bl = q >> 4;        // 0..31
    const int gb = b0 + bl;
    const int pbase = LSWZ(r, bl);   // k-invariant part of the read index
    float acc[8] = {0.f, 0.f, 0.f, 0.f, 0.f, 0.f, 0.f, 0.f};
    #pragma unroll
    for (int c = 0; c < CCLS; ++c) {
      const float pc = p[(size_t)gb * CCLS + c];
      float l[8];
      #pragma unroll
      for (int d = 0; d < 8; ++d) l[d] = lds[pbase + (c * 8 + d) * (RT * BT)];
      float mx = l[0];
      #pragma unroll
      for (int d = 1; d < 8; ++d) mx = fmaxf(mx, l[d]);
      float e[8], s = 0.f;
      #pragma unroll
      for (int d = 0; d < 8; ++d) { e[d] = __expf(l[d] - mx); s += e[d]; }
      const float coef = pc / s;
      #pragma unroll
      for (int d = 0; d < 8; ++d) acc[d] += coef * e[d];
    }
    #pragma unroll
    for (int d = 0; d < 8; ++d)
      out[(size_t)gb * (CCLS * RDIM) + d * RDIM + r0 + r] = acc[d];
  }
}

// ---------------------------------------------------------------------------
extern "C" void kernel_launch(void* const* d_in, const int* in_sizes, int n_in,
                              void* d_out, int out_size, void* d_ws, size_t ws_size,
                              hipStream_t stream) {
  const float* x    = (const float*)d_in[0];   // (16384, 128)
  const float* p    = (const float*)d_in[1];   // (16384, 8)
  const float* W    = (const float*)d_in[2];   // (128, 64, 64)
  const float* bias = (const float*)d_in[3];   // (8, 8, 64)
  float* out = (float*)d_out;                  // (16384, 8, 64)

  _Float16* Wt = (_Float16*)d_ws;              // needs 4096*128*2 = 1 MB

  conv_w_kernel<<<(FDIM * NCOLS) / 256, 256, 0, stream>>>(W, Wt);

  dim3 grid(RDIM / RT, BDIM / BT);             // (4, 512)
  fused_kernel<<<grid, 256, 0, stream>>>(x, p, Wt, bias, out);
}
